// Decoder_62586263437796
// MI455X (gfx1250) — matrix operand order repaired and checked by reading, not compiled, not run on hardware
//
#include <hip/hip_runtime.h>
#include <hip/hip_bf16.h>
#include <math.h>

// ---------------------------------------------------------------------------
// Tacotron-style decoder for MI455X (gfx1250, wave32, WMMA).
// - All GEMMs: V_WMMA_F32_16X16X32_F16 (f16 in, f32 acc), weights pre-swizzled
//   into the ISA wave32 B-matrix VGPR layout (contiguous 32B per lane/K-step).
// - Recurrent LSTM steps: fused GEMM + gate nonlinearity (4 gate accumulators
//   per wave sharing one A load), f16 activations staged into LDS with
//   GLOBAL_LOAD_ASYNC_TO_LDS_B128 + s_wait_asynccnt (ASYNCcnt path).
// - h1/h2 are ping-pong buffers (fused kernel reads prev h, writes new h).
// ---------------------------------------------------------------------------

typedef __attribute__((ext_vector_type(16))) _Float16 v16h;
typedef __attribute__((ext_vector_type(8)))  float    v8f;

#define BB   32     // batch
#define TT   128    // tokens
#define FF   80     // mel dims
#define SS   128    // speaker
#define PP   256    // prenet
#define HH   1024   // lstm hidden
#define EE   512    // token embed
#define NN   300    // decode steps
#define HAA  128    // attention hidden
#define KK   9      // conv taps
#define WLL  15     // window
#define PADD 4      // (K-1)/2
#define CUML (TT + 2 * PADD)   // 136

#define GEMM_K_MAX 2688        // max K (lstm2: 1024+512+128+1024)

struct Seg {
    const void* ptr;
    int n;       // K columns contributed by this segment (0 = unused)
    int ld;      // row stride (elements)
    int isHalf;  // 1 -> _Float16 data, 0 -> float
};

__device__ inline float seg_read(const Seg& s, int m, int k) {
    return s.isHalf ? (float)((const _Float16*)s.ptr)[(size_t)m * s.ld + k]
                    : ((const float*)s.ptr)[(size_t)m * s.ld + k];
}

__device__ inline float sigmoidf_(float x) { return 1.0f / (1.0f + __expf(-x)); }

// Gather one 16x32 A-tile (ISA A-matrix layout) from an LDS row panel.
__device__ inline v16h load_a_tile(const _Float16* arow, int kb) {
    v16h a;
    #pragma unroll
    for (int j = 0; j < 16; ++j) a[j] = arow[kb + ((j >> 3) << 4) + (j & 7)];
    return a;
}

// ---------------------------------------------------------------------------
// Weight swizzle: pack W (OUT x (Ka|Kb) concat along K) into f16 B-tiles:
// layout [nt][kt][lane][j], K = kt*32 + (lane>=16?8:0) + (j>=8?16:0) + (j&7), N = nt*16+(lane&15)
// ---------------------------------------------------------------------------
__global__ void swizzle_cat2(const float* __restrict__ Wa, int Ka,
                             const float* __restrict__ Wb, int Kb,
                             int OUT, int Ktiles, int Ntiles,
                             _Float16* __restrict__ dst)
{
    size_t total = (size_t)Ntiles * Ktiles * 512;
    for (size_t idx = blockIdx.x * (size_t)blockDim.x + threadIdx.x;
         idx < total; idx += (size_t)gridDim.x * blockDim.x) {
        int    j    = (int)(idx & 15);
        int    lane = (int)((idx >> 4) & 31);
        size_t tile = idx >> 9;
        int    kt   = (int)(tile % Ktiles);
        int    nt   = (int)(tile / Ktiles);
        int n = nt * 16 + (lane & 15);
        int k = kt * 32 + ((lane >> 4) << 3) + ((j >> 3) << 4) + (j & 7);
        float v = 0.0f;
        if (n < OUT) {
            if (k < Ka)                 v = Wa[(size_t)n * Ka + k];
            else if (Wb && k < Ka + Kb) v = Wb[(size_t)n * Kb + (k - Ka)];
        }
        dst[idx] = (_Float16)v;
    }
}

// ---------------------------------------------------------------------------
// Generic Y = act( [s0|s1|s2|s3] @ W^T + bias0 + bias1 )  (setup/batched GEMMs)
// Block = 128 threads (4 waves); block computes 16(M) x 64(N).
// ---------------------------------------------------------------------------
__global__ __launch_bounds__(128) void gemm_cat4(
    Seg s0, Seg s1, Seg s2, Seg s3,
    const _Float16* __restrict__ Wsw, int Ktiles,
    const float* __restrict__ bias0, const float* __restrict__ bias1,
    float* __restrict__ Y, int ldy, int M, int Nout, int act)
{
    __shared__ _Float16 ldsX[16 * GEMM_K_MAX];
    const int Ksum = Ktiles * 32;
    const int row0 = blockIdx.y * 16;

    for (int idx = threadIdx.x; idx < 16 * Ksum; idx += 128) {
        int r = idx / Ksum;
        int k = idx - r * Ksum;
        int m = row0 + r;
        float v = 0.0f;
        if (m < M) {
            int kk = k;
            if (kk < s0.n) v = seg_read(s0, m, kk);
            else {
                kk -= s0.n;
                if (kk < s1.n) v = seg_read(s1, m, kk);
                else {
                    kk -= s1.n;
                    if (kk < s2.n) v = seg_read(s2, m, kk);
                    else {
                        kk -= s2.n;
                        if (kk < s3.n) v = seg_read(s3, m, kk);
                    }
                }
            }
        }
        ldsX[idx] = (_Float16)v;
    }
    __syncthreads();

    const int wave = threadIdx.x >> 5;
    const int lane = threadIdx.x & 31;
    const int n0   = blockIdx.x * 64 + wave * 16;
    if (n0 >= Nout) return;

    v8f c = {};
    const _Float16* bp   = Wsw + ((size_t)(n0 >> 4) * Ktiles) * 512 + lane * 16;
    const _Float16* arow = ldsX + (size_t)(lane & 15) * Ksum;
    const int kb = (lane >> 4) << 3;

    for (int kt = 0; kt < Ktiles; ++kt) {
        v16h a = load_a_tile(arow + kt * 32, kb);
        v16h b = *(const v16h*)(bp + (size_t)kt * 512);
        c = __builtin_amdgcn_wmma_f32_16x16x32_f16(
                false, a, false, b, (short)0, c, false, false);
    }

    const int n = n0 + (lane & 15);
    if (n < Nout) {
        float bsum = 0.0f;
        if (bias0) bsum += bias0[n];
        if (bias1) bsum += bias1[n];
        const int mhi = (lane >> 4) << 3;
        #pragma unroll
        for (int i = 0; i < 8; ++i) {
            int m = row0 + mhi + i;
            if (m < M) {
                float v = c[i] + bsum;
                if (act == 1) v = fmaxf(v, 0.0f);
                Y[(size_t)m * ldy + n] = v;
            }
        }
    }
}

// ---------------------------------------------------------------------------
// Fused LSTM step: gates = [s0|s1|s2|s3](f16) @ (Wih|Whh)^T + bih + bhh, then
// c' = sig(f)*c + sig(i)*tanh(g), h' = sig(o)*tanh(c').
// Each wave holds 4 gate accumulators (i,f,g,o for the same 16 hidden cols),
// sharing one A load per K-step.  A-panel staged via async-to-LDS (ASYNCcnt).
// Grid: (HH/64, 2) x 128 threads.  M = 32 exactly.
// ---------------------------------------------------------------------------
__global__ __launch_bounds__(128) void lstm_step_fused(
    Seg s0, Seg s1, Seg s2, Seg s3,            // all f16, K multiple of 8
    const _Float16* __restrict__ Wsw, int Ktiles,
    const float* __restrict__ bias0, const float* __restrict__ bias1,
    float* __restrict__ cbuf,                  // 32 x HH f32 cell state
    _Float16* __restrict__ hOut, int ldh,      // new hidden (f16)
    _Float16* __restrict__ hOut2, int ldh2)    // optional 2nd copy (Xfinal row)
{
    __shared__ _Float16 ldsX[16 * GEMM_K_MAX];
    const int Ksum = Ktiles * 32;
    const int row0 = blockIdx.y * 16;
    const int ckPerRow = Ksum >> 3;            // 16-byte chunks per row

    // ---- async copy 16 rows of concatenated f16 activations into LDS ----
    for (int idx = threadIdx.x; idx < 16 * ckPerRow; idx += 128) {
        int r = idx / ckPerRow;
        int k = (idx - r * ckPerRow) << 3;
        int m = row0 + r;
        const _Float16* src;
        int kk = k;
        if (kk < s0.n) src = (const _Float16*)s0.ptr + (size_t)m * s0.ld + kk;
        else {
            kk -= s0.n;
            if (kk < s1.n) src = (const _Float16*)s1.ptr + (size_t)m * s1.ld + kk;
            else {
                kk -= s1.n;
                if (kk < s2.n) src = (const _Float16*)s2.ptr + (size_t)m * s2.ld + kk;
                else { kk -= s2.n; src = (const _Float16*)s3.ptr + (size_t)m * s3.ld + kk; }
            }
        }
        unsigned ldsOff = (unsigned)(uintptr_t)(&ldsX[r * Ksum + k]);
        unsigned long long ga = (unsigned long long)(uintptr_t)src;
        asm volatile("global_load_async_to_lds_b128 %0, %1, off"
                     :: "v"(ldsOff), "v"(ga) : "memory");
    }
    asm volatile("s_wait_asynccnt 0x0" ::: "memory");
    __syncthreads();

    const int wave = threadIdx.x >> 5;
    const int lane = threadIdx.x & 31;
    const int n0   = blockIdx.x * 64 + wave * 16;   // hidden column tile

    v8f acc[4] = {{}, {}, {}, {}};
    const _Float16* arow = ldsX + (size_t)(lane & 15) * Ksum;
    const int kb = (lane >> 4) << 3;
    const size_t tstride = (size_t)Ktiles * 512;
    const _Float16* bp0 = Wsw + (size_t)(n0 >> 4) * tstride + lane * 16;

    for (int kt = 0; kt < Ktiles; ++kt) {
        v16h a = load_a_tile(arow + kt * 32, kb);
        #pragma unroll
        for (int g = 0; g < 4; ++g) {
            v16h b = *(const v16h*)(bp0 + (size_t)(g * 64) * tstride + (size_t)kt * 512);
            acc[g] = __builtin_amdgcn_wmma_f32_16x16x32_f16(
                        false, a, false, b, (short)0, acc[g], false, false);
        }
    }

    const int n = n0 + (lane & 15);
    float bs[4];
    #pragma unroll
    for (int g = 0; g < 4; ++g) bs[g] = bias0[g * HH + n] + bias1[g * HH + n];
    const int mhi = (lane >> 4) << 3;
    #pragma unroll
    for (int i = 0; i < 8; ++i) {
        int m = row0 + mhi + i;
        float gi = acc[0][i] + bs[0];
        float gf = acc[1][i] + bs[1];
        float gg = acc[2][i] + bs[2];
        float go = acc[3][i] + bs[3];
        float cv = cbuf[(size_t)m * HH + n];
        cv = sigmoidf_(gf) * cv + sigmoidf_(gi) * tanhf(gg);
        float hv = sigmoidf_(go) * tanhf(cv);
        cbuf[(size_t)m * HH + n] = cv;
        hOut[(size_t)m * ldh + n] = (_Float16)hv;
        if (hOut2) hOut2[(size_t)m * ldh2 + n] = (_Float16)hv;
    }
}

// ---------------------------------------------------------------------------
// Initial state: ctx0 (f16), cum0(|init_ca|), zero h(f16)/c(f32), speaker f16.
// ---------------------------------------------------------------------------
__global__ void init_state(const float* __restrict__ st2, // ld 608
                           const float* __restrict__ speaker,
                           _Float16* __restrict__ ctxh, float* __restrict__ cum,
                           int* __restrict__ ws,
                           _Float16* __restrict__ h1h0, float* __restrict__ c1,
                           _Float16* __restrict__ h2h0, float* __restrict__ c2,
                           _Float16* __restrict__ spkHalf)
{
    int tid = blockIdx.x * blockDim.x + threadIdx.x;
    if (tid < BB * EE) {
        int b = tid >> 9, e = tid & 511;
        ctxh[tid] = (_Float16)st2[b * 608 + FF + 1 + e];
    }
    int t2 = tid - BB * EE;
    if (t2 >= 0 && t2 < BB * CUML) {
        int b = t2 / CUML, i = t2 % CUML;
        cum[t2] = (i < PADD) ? fabsf(st2[b * 608 + FF]) : 0.0f;
    }
    int t3 = t2 - BB * CUML;
    if (t3 >= 0 && t3 < BB) ws[t3] = 0;
    int t4 = t3 - BB;
    if (t4 >= 0 && t4 < BB * HH) {
        c1[t4] = 0.f; c2[t4] = 0.f;
        h1h0[t4] = (_Float16)0.f; h2h0[t4] = (_Float16)0.f;
    }
    int t5 = t4 - BB * HH;
    if (t5 >= 0 && t5 < BB * SS) spkHalf[t5] = (_Float16)speaker[t5];
}

// ---------------------------------------------------------------------------
// Prenet input pack (f16, K padded 208->224) and speaker columns of final-X.
// ---------------------------------------------------------------------------
__global__ void prepack(const float* __restrict__ st2,
                        const float* __restrict__ target_frames,
                        const float* __restrict__ speaker,
                        _Float16* __restrict__ Xpre,     // (N*B) x 224
                        _Float16* __restrict__ Xfinal)   // (N*B) x 1664
{
    const size_t totalA = (size_t)NN * BB * 224;
    const size_t totalB = (size_t)NN * BB * SS;
    for (size_t idx = blockIdx.x * (size_t)blockDim.x + threadIdx.x;
         idx < totalA + totalB; idx += (size_t)gridDim.x * blockDim.x) {
        if (idx < totalA) {
            int r = (int)(idx / 224), cidx = (int)(idx % 224);
            int t = r / BB, b = r % BB;
            float v = 0.0f;
            if (cidx < FF)
                v = (t == 0) ? st2[b * 608 + cidx]
                             : target_frames[((size_t)(t - 1) * BB + b) * FF + cidx];
            else if (cidx < FF + SS)
                v = speaker[b * SS + (cidx - FF)];
            Xpre[idx] = (_Float16)v;
        } else {
            size_t i2 = idx - totalA;
            int r = (int)(i2 / SS), cidx = (int)(i2 % SS);
            int b = r % BB;
            Xfinal[(size_t)r * 1664 + HH + EE + cidx] = (_Float16)speaker[b * SS + cidx];
        }
    }
}

// ---------------------------------------------------------------------------
// Prenet relu + LayerNorm(256) -> f16.  One block (256 threads) per row.
// ---------------------------------------------------------------------------
__global__ __launch_bounds__(256) void prenet_ln(const float* __restrict__ lin,
                                                 const float* __restrict__ g,
                                                 const float* __restrict__ bp,
                                                 _Float16* __restrict__ out)
{
    __shared__ float red[256];
    const int row = blockIdx.x;
    float x = fmaxf(lin[(size_t)row * PP + threadIdx.x], 0.0f);
    red[threadIdx.x] = x;
    __syncthreads();
    for (int s = 128; s > 0; s >>= 1) {
        if (threadIdx.x < s) red[threadIdx.x] += red[threadIdx.x + s];
        __syncthreads();
    }
    float m = red[0] / 256.0f;
    __syncthreads();
    float d = x - m;
    red[threadIdx.x] = d * d;
    __syncthreads();
    for (int s = 128; s > 0; s >>= 1) {
        if (threadIdx.x < s) red[threadIdx.x] += red[threadIdx.x + s];
        __syncthreads();
    }
    float v = red[0] / 256.0f;
    float y = d * rsqrtf(v + 1e-5f) * g[threadIdx.x] + bp[threadIdx.x];
    out[(size_t)row * PP + threadIdx.x] = (_Float16)y;
}

// ---------------------------------------------------------------------------
// Windowed location-sensitive attention.  One block of 128 threads per batch.
// ---------------------------------------------------------------------------
__global__ __launch_bounds__(128) void attention_kernel(
    int t,
    const _Float16* __restrict__ h1,           // current h1 (f16, ld HH)
    const float* __restrict__ tokens,          // (T,B,E)
    const unsigned char* __restrict__ tokens_mask, // (B,T)
    const int* __restrict__ num_tokens,
    const float* __restrict__ conv_w, const float* __restrict__ conv_b,
    const float* __restrict__ Wq, const float* __restrict__ bq,
    const float* __restrict__ ln_g, const float* __restrict__ ln_b,
    const float* __restrict__ vatt,
    _Float16* __restrict__ ctxh, float* __restrict__ cum, int* __restrict__ ws,
    _Float16* __restrict__ XfinalRow,  // Xfinal + t*B*1664
    float* __restrict__ outAlign,
    float* __restrict__ outWss)
{
    const int b = blockIdx.x;
    const int h = threadIdx.x;
    __shared__ float cumwin[WLL + 2 * PADD];
    __shared__ float tq[WLL][HAA + 1];
    __shared__ float sc[WLL], aw[WLL];
    __shared__ int wsold_s, wsnew_s;

    if (h == 0) wsold_s = ws[b];
    __syncthreads();
    const int wsold = wsold_s;
    if (h < WLL + 2 * PADD) cumwin[h] = cum[b * CUML + wsold + h];
    __syncthreads();

    float q = bq[h];
    const _Float16* h1b = h1 + (size_t)b * HH;
    const float* wqh = Wq + (size_t)h * HH;
    for (int k = 0; k < HH; ++k) q += (float)h1b[k] * wqh[k];

    const float* cwh = conv_w + h * KK;
    const float cb = conv_b[h];
    for (int w = 0; w < WLL; ++w) {
        float f = cb;
        #pragma unroll
        for (int k = 0; k < KK; ++k) f += cumwin[w + k] * cwh[k];
        tq[w][h] = tanhf(f + q);
    }
    __syncthreads();

    if (h < WLL) {
        float m = 0.0f;
        for (int j = 0; j < HAA; ++j) m += tq[h][j];
        m /= (float)HAA;
        float v = 0.0f;
        for (int j = 0; j < HAA; ++j) { float d = tq[h][j] - m; v += d * d; }
        v /= (float)HAA;
        float inv = rsqrtf(v + 1e-5f);
        float s = 0.0f;
        for (int j = 0; j < HAA; ++j)
            s += ((tq[h][j] - m) * inv * ln_g[j] + ln_b[j]) * vatt[j];
        int tok = wsold + h;
        sc[h] = (tokens_mask[b * TT + tok] != 0) ? s : -1e9f;
    }
    __syncthreads();

    if (h == 0) {
        float mx = sc[0];
        int am = 0;
        for (int w = 1; w < WLL; ++w)
            if (sc[w] > mx) { mx = sc[w]; am = w; }
        float sum = 0.0f;
        for (int w = 0; w < WLL; ++w) { float e = __expf(sc[w] - mx); aw[w] = e; sum += e; }
        float inv = 1.0f / sum;
        for (int w = 0; w < WLL; ++w) aw[w] *= inv;
        int wmax = num_tokens[b] - WLL;
        if (wmax < 0) wmax = 0;
        int wn = wsold + am - WLL / 2;
        if (wn < 0) wn = 0;
        if (wn > wmax) wn = wmax;
        wsnew_s = wn;
        outWss[t * BB + b] = (float)wn;
    }
    __syncthreads();

    for (int e = h; e < EE; e += HAA) {
        float s = 0.0f;
        for (int w = 0; w < WLL; ++w)
            s += aw[w] * tokens[((size_t)(wsold + w) * BB + b) * EE + e];
        ctxh[b * EE + e] = (_Float16)s;
        XfinalRow[(size_t)b * 1664 + HH + e] = (_Float16)s;
    }

    float* al = outAlign + (size_t)(t * BB + b) * TT;
    al[h] = 0.0f;
    __syncthreads();
    if (h < WLL) {
        al[wsold + h] = aw[h];
        cum[b * CUML + wsold + h + PADD] += aw[h];
    }
    if (h == 0) ws[b] = wsnew_s;
}

// ---------------------------------------------------------------------------
// stop = h2 @ stop_W^T + stop_b  (reads f16 h2 from final-X buffer)
// ---------------------------------------------------------------------------
__global__ void stop_kernel(const _Float16* __restrict__ Xf,
                            const float* __restrict__ sw,
                            const float* __restrict__ sb,
                            float* __restrict__ out)
{
    int row = blockIdx.x * blockDim.x + threadIdx.x;
    if (row >= NN * BB) return;
    const _Float16* x = Xf + (size_t)row * 1664;
    float s = sb[0];
    for (int k = 0; k < HH; ++k) s += (float)x[k] * sw[k];
    out[row] = s;
}

// ---------------------------------------------------------------------------
// Host orchestration
// ---------------------------------------------------------------------------
extern "C" void kernel_launch(void* const* d_in, const int* in_sizes, int n_in,
                              void* d_out, int out_size, void* d_ws, size_t ws_size,
                              hipStream_t stream)
{
    const float*         tokens       = (const float*)d_in[0];
    const unsigned char* tokens_mask  = (const unsigned char*)d_in[1];
    const int*           num_tokens   = (const int*)d_in[2];
    const float*         speaker      = (const float*)d_in[3];
    const float*         target_frames= (const float*)d_in[4];
    const float*         init_W1      = (const float*)d_in[5];
    const float*         init_b1      = (const float*)d_in[6];
    const float*         init_W2      = (const float*)d_in[7];
    const float*         init_b2      = (const float*)d_in[8];
    const float*         pre_W        = (const float*)d_in[9];
    const float*         pre_b        = (const float*)d_in[10];
    const float*         pre_ln_g     = (const float*)d_in[11];
    const float*         pre_ln_b     = (const float*)d_in[12];
    const float*         l1_Wih       = (const float*)d_in[13];
    const float*         l1_Whh       = (const float*)d_in[14];
    const float*         l1_bih       = (const float*)d_in[15];
    const float*         l1_bhh       = (const float*)d_in[16];
    const float*         att_conv_w   = (const float*)d_in[17];
    const float*         att_conv_b   = (const float*)d_in[18];
    const float*         att_Wq       = (const float*)d_in[19];
    const float*         att_bq       = (const float*)d_in[20];
    const float*         att_ln_g     = (const float*)d_in[21];
    const float*         att_ln_b     = (const float*)d_in[22];
    const float*         att_v        = (const float*)d_in[23];
    const float*         l2_Wih       = (const float*)d_in[24];
    const float*         l2_Whh       = (const float*)d_in[25];
    const float*         l2_bih       = (const float*)d_in[26];
    const float*         l2_bhh       = (const float*)d_in[27];
    const float*         out_W        = (const float*)d_in[28];
    const float*         out_b        = (const float*)d_in[29];
    const float*         stop_W       = (const float*)d_in[30];
    const float*         stop_b       = (const float*)d_in[31];

    float* outFrames = (float*)d_out;                           // N*B*F
    float* outStop   = outFrames + (size_t)NN * BB * FF;        // N*B
    float* outAlign  = outStop + (size_t)NN * BB;               // N*B*T
    float* outWss    = outAlign + (size_t)NN * BB * TT;         // N*B

    size_t off = 0;
    auto alloc = [&](size_t bytes) -> void* {
        void* p = (char*)d_ws + off;
        off += (bytes + 255) & ~(size_t)255;
        return p;
    };
    _Float16* w1sw   = (_Float16*)alloc((size_t)640 * 640 * 2);
    _Float16* w2sw   = (_Float16*)alloc((size_t)608 * 640 * 2);
    _Float16* presw  = (_Float16*)alloc((size_t)256 * 224 * 2);
    _Float16* l1sw   = (_Float16*)alloc((size_t)4096 * 1920 * 2);
    _Float16* l2sw   = (_Float16*)alloc((size_t)4096 * 2688 * 2);
    _Float16* outsw  = (_Float16*)alloc((size_t)80 * 1664 * 2);
    float*    st1    = (float*)alloc((size_t)BB * 640 * 4);
    float*    st2    = (float*)alloc((size_t)BB * 608 * 4);
    _Float16* ctxh   = (_Float16*)alloc((size_t)BB * EE * 2);
    float*    cum    = (float*)alloc((size_t)BB * CUML * 4);
    int*      wsbuf  = (int*)alloc((size_t)BB * 4);
    _Float16* h1buf0 = (_Float16*)alloc((size_t)BB * HH * 2);   // ping-pong
    _Float16* h1buf1 = (_Float16*)alloc((size_t)BB * HH * 2);
    _Float16* h2buf0 = (_Float16*)alloc((size_t)BB * HH * 2);
    _Float16* h2buf1 = (_Float16*)alloc((size_t)BB * HH * 2);
    float*    c1     = (float*)alloc((size_t)BB * HH * 4);
    float*    c2     = (float*)alloc((size_t)BB * HH * 4);
    _Float16* spkH   = (_Float16*)alloc((size_t)BB * SS * 2);
    _Float16* Xpre   = (_Float16*)alloc((size_t)NN * BB * 224 * 2);
    float*    pnlin  = (float*)alloc((size_t)NN * BB * PP * 4);
    _Float16* pnf16  = (_Float16*)alloc((size_t)NN * BB * PP * 2);
    _Float16* Xfinal = (_Float16*)alloc((size_t)NN * BB * 1664 * 2);

    // ---- weight swizzles ----
    auto swzGrid = [](size_t elems) { return (unsigned)((elems + 255) / 256); };
    swizzle_cat2<<<swzGrid((size_t)40 * 20 * 512), 256, 0, stream>>>(
        init_W1, 640, nullptr, 0, 640, 20, 40, w1sw);
    swizzle_cat2<<<swzGrid((size_t)38 * 20 * 512), 256, 0, stream>>>(
        init_W2, 640, nullptr, 0, 593, 20, 38, w2sw);
    swizzle_cat2<<<swzGrid((size_t)16 * 7 * 512), 256, 0, stream>>>(
        pre_W, 208, nullptr, 0, 256, 7, 16, presw);
    swizzle_cat2<<<swzGrid((size_t)256 * 60 * 512), 256, 0, stream>>>(
        l1_Wih, 896, l1_Whh, 1024, 4096, 60, 256, l1sw);
    swizzle_cat2<<<swzGrid((size_t)256 * 84 * 512), 256, 0, stream>>>(
        l2_Wih, 1664, l2_Whh, 1024, 4096, 84, 256, l2sw);
    swizzle_cat2<<<swzGrid((size_t)5 * 52 * 512), 256, 0, stream>>>(
        out_W, 1664, nullptr, 0, 80, 52, 5, outsw);

    Seg none = { nullptr, 0, 0, 0 };

    // ---- init: st1 = relu([speaker|tokens0] @ W1^T + b1) ----
    {
        Seg sA = { speaker, SS, SS, 0 };
        Seg sB = { tokens, EE, EE, 0 };   // tokens[0] = first B*E floats
        gemm_cat4<<<dim3(10, 2), 128, 0, stream>>>(
            sA, sB, none, none, w1sw, 20, init_b1, nullptr, st1, 640, BB, 640, 1);
    }
    // ---- st2 = st1 @ W2^T + b2 ----
    {
        Seg sA = { st1, 640, 640, 0 };
        gemm_cat4<<<dim3(10, 2), 128, 0, stream>>>(
            sA, none, none, none, w2sw, 20, init_b2, nullptr, st2, 608, BB, 593, 0);
    }
    init_state<<<226, 256, 0, stream>>>(st2, speaker, ctxh, cum, wsbuf,
                                        h1buf0, c1, h2buf0, c2, spkH);
    prepack<<<4096, 256, 0, stream>>>(st2, target_frames, speaker, Xpre, Xfinal);

    // ---- prenet GEMM + relu/LN ----
    {
        Seg sA = { Xpre, 224, 224, 1 };
        gemm_cat4<<<dim3(4, 600), 128, 0, stream>>>(
            sA, none, none, none, presw, 7, pre_b, nullptr, pnlin, PP, NN * BB, PP, 0);
    }
    prenet_ln<<<NN * BB, 256, 0, stream>>>(pnlin, pre_ln_g, pre_ln_b, pnf16);

    // ---- 300 autoregressive steps ----
    _Float16* h1ping[2] = { h1buf0, h1buf1 };
    _Float16* h2ping[2] = { h2buf0, h2buf1 };
    for (int t = 0; t < NN; ++t) {
        _Float16* XfRow = Xfinal + (size_t)t * BB * 1664;
        _Float16* h1in  = h1ping[t & 1];
        _Float16* h1out = h1ping[(t + 1) & 1];
        _Float16* h2in  = h2ping[t & 1];
        _Float16* h2out = h2ping[(t + 1) & 1];

        // LSTM1: [pn_t | ctx | spk | h1] -> h1', c1'
        {
            Seg s0 = { pnf16 + (size_t)t * BB * PP, PP, PP, 1 };
            Seg s1 = { ctxh, EE, EE, 1 };
            Seg s2 = { spkH, SS, SS, 1 };
            Seg s3 = { h1in, HH, HH, 1 };
            lstm_step_fused<<<dim3(16, 2), 128, 0, stream>>>(
                s0, s1, s2, s3, l1sw, 60, l1_bih, l1_bhh,
                c1, h1out, HH, nullptr, 0);
        }

        attention_kernel<<<BB, HAA, 0, stream>>>(
            t, h1out, tokens, tokens_mask, num_tokens,
            att_conv_w, att_conv_b, att_Wq, att_bq, att_ln_g, att_ln_b, att_v,
            ctxh, cum, wsbuf, XfRow, outAlign, outWss);

        // LSTM2: [h1' | ctx' | spk | h2] -> h2', c2'  (h2' also into Xfinal row)
        {
            Seg s0 = { h1out, HH, HH, 1 };
            Seg s1 = { ctxh, EE, EE, 1 };
            Seg s2 = { spkH, SS, SS, 1 };
            Seg s3 = { h2in, HH, HH, 1 };
            lstm_step_fused<<<dim3(16, 2), 128, 0, stream>>>(
                s0, s1, s2, s3, l2sw, 84, l2_bih, l2_bhh,
                c2, h2out, HH, XfRow, 1664);
        }
    }

    // ---- frames = [h2 | ctx | spk] @ out_W^T + out_b ----
    {
        Seg sA = { Xfinal, 1664, 1664, 1 };
        gemm_cat4<<<dim3(2, 600), 128, 0, stream>>>(
            sA, none, none, none, outsw, 52, out_b, nullptr, outFrames, FF, NN * BB, FF, 0);
    }
    // ---- stop logits ----
    stop_kernel<<<(NN * BB + 255) / 256, 256, 0, stream>>>(Xfinal, stop_W, stop_b, outStop);

    (void)in_sizes; (void)n_in; (void)out_size; (void)ws_size;
}
